// ConvLSTM_87162066305211
// MI455X (gfx1250) — matrix operand order repaired and checked by reading, not compiled, not run on hardware
//
#include <hip/hip_runtime.h>
#include <hip/hip_bf16.h>

// ---------------------------------------------------------------------------
// ConvLSTM x3 + BN + MaxPool + GAP + Dense + Softmax for MI455X (gfx1250).
// Matrix work on V_WMMA_F32_16X16X32_BF16 (wave32); weights staged in LDS
// via GLOBAL_LOAD_ASYNC_TO_LDS_B128 (ASYNCcnt), B fragments double-buffered.
// ---------------------------------------------------------------------------

typedef __attribute__((ext_vector_type(16))) __bf16 v16bf;
typedef __attribute__((ext_vector_type(8)))  __bf16 v8bf;
typedef __attribute__((ext_vector_type(8)))  float  v8f;
typedef int v4i __attribute__((vector_size(16)));
typedef __attribute__((address_space(1))) v4i gv4i;   // global int4
typedef __attribute__((address_space(3))) v4i lv4i;   // LDS int4

#if defined(__has_builtin)
#if __has_builtin(__builtin_amdgcn_global_load_async_to_lds_b128) && \
    __has_builtin(__builtin_amdgcn_s_wait_asynccnt)
#define USE_ASYNC_LDS 1
#endif
#endif
#ifndef USE_ASYNC_LDS
#define USE_ASYNC_LDS 0
#endif

__device__ __forceinline__ float hard_sig(float z) {
    return fminf(fmaxf(z * (1.0f / 6.0f) + 0.5f, 0.0f), 1.0f);
}

// ---------------------------------------------------------------------------
// Pack conv weights (kh,kw,32,128) f32 -> bf16 in WMMA B-fragment lane order.
// B is 32(K) x 16(N) per tile: lane half h, elem e -> K = 8h+e (e<8),
// K = 16+8h+(e-8) (e>=8); n = lane&15; tile j covers channels 16j..16j+15.
// dst[((tap*8 + j)*32 + lane)*16 + e]   (8 KB per tap, tap-major)
// ---------------------------------------------------------------------------
__global__ __launch_bounds__(256) void pack_b_kern(
    const float* __restrict__ src, __bf16* __restrict__ dst, int ntaps)
{
    int idx = blockIdx.x * 256 + threadIdx.x;
    int total = ntaps * 8 * 32 * 16;
    if (idx >= total) return;
    int e    = idx & 15;
    int rest = idx >> 4;
    int lane = rest & 31; rest >>= 5;
    int j    = rest & 7;
    int tap  = rest >> 3;
    int k    = (e & 7) + 8 * (lane >> 4) + 16 * (e >> 3);
    int ch   = 16 * j + (lane & 15);
    dst[idx] = (__bf16)src[((size_t)tap * 32 + k) * 128 + ch];
}

// ---------------------------------------------------------------------------
// One ConvLSTM timestep. Each wave computes a 16-pixel x 128-channel z-tile
// via WMMA over kernel taps (K=32 per tap), then the LSTM cell update.
//   z = conv(x_t, Wk) + conv(h_{t-1}, Uk) + b
//   i,f,o = hardsig ; g = relu ; c' = f*c + i*g ; h' = o*relu(c')
// Layer1 (X1=true): Cin=1 x-conv done as VALU preamble on the accumulators.
// One ky-row of packed B taps (KH * 8KB) is staged in LDS per iteration and
// shared by all 8 waves of the block.
// ---------------------------------------------------------------------------
template <int H, int W, int KH, bool X1>
__global__ __launch_bounds__(256) void lstm_step_kern(
    const __bf16* __restrict__ xin,   // layers 2/3: pooled seq (B,T,H,W,32)
    const float*  __restrict__ x1,    // layer 1: raw x (B,T,H,W,1)
    const float*  __restrict__ w1,    // layer 1: k1 as (KH*KH, 128) f32
    const __bf16* __restrict__ packW, // layers 2/3: packed Wk
    const __bf16* __restrict__ packU, // packed Uk
    const float*  __restrict__ bias,  // (128)
    const __bf16* __restrict__ hin,   // (B*H*W, 32)
    __bf16*       __restrict__ hout,  // (B*H*W, 32)
    float*        __restrict__ cbuf,  // (32, B*H*W)
    int t)
{
    constexpr int Tn = 16, PAD = KH / 2, M = 16 * H * W;
    __shared__ __align__(16) char smem[KH * 8192];   // one ky-row of B taps

    const int wave  = threadIdx.x >> 5;
    const int lane  = threadIdx.x & 31;
    const int tile  = blockIdx.x * 8 + wave;
    const int pbase = tile << 4;                 // 16 consecutive x in one row
    const int b     = pbase / (H * W);
    const int rem   = pbase - b * (H * W);
    const int y     = rem / W;
    const int x0    = rem - y * W;
    const int n     = lane & 15;                 // N column of this lane
    const int hi    = lane >> 4;                 // K-half select for A/B
    const int cb    = hi << 3;                   // A channel base {0,8}

    // 8 accumulator tiles cover channels 16j+n; rows m = 8*hi + r.
    v8f acc[8];
#pragma unroll
    for (int j = 0; j < 8; ++j) {
        float bv = bias[16 * j + n];
#pragma unroll
        for (int r = 0; r < 8; ++r) acc[j][r] = bv;
    }

    if constexpr (X1) {
        // x-conv, Cin = 1: plain FMAs into the accumulators (~3% of FLOPs).
        for (int ky = 0; ky < KH; ++ky) {
            int yy = y + ky - PAD;
            if (yy < 0 || yy >= H) continue;     // wave-uniform
            for (int kx = 0; kx < KH; ++kx) {
                int tap = ky * KH + kx;
                float wv[8];
#pragma unroll
                for (int j = 0; j < 8; ++j) wv[j] = w1[tap * 128 + 16 * j + n];
#pragma unroll
                for (int r = 0; r < 8; ++r) {
                    int xx = x0 + (hi << 3) + r + kx - PAD;
                    float xv = 0.0f;
                    if (xx >= 0 && xx < W)
                        xv = x1[(((size_t)(b * Tn + t) * H + yy) * W) + xx];
#pragma unroll
                    for (int j = 0; j < 8; ++j)
                        acc[j][r] = fmaf(xv, wv[j], acc[j][r]);
                }
            }
        }
    }

    // WMMA passes: pass 0 = x-conv (layers 2/3 only), pass 1 = h-conv.
    const int npass = X1 ? 1 : 2;
    for (int pass = 0; pass < npass; ++pass) {
        const __bf16* img;
        const __bf16* pk;
        if (X1 || pass == 1) { img = hin + (size_t)b * (H * W) * 32; pk = packU; }
        else { img = xin + (size_t)(b * Tn + t) * (H * W) * 32;      pk = packW; }

        for (int ky = 0; ky < KH; ++ky) {
            // ---- stage this ky-row of packed B taps into LDS (all waves) ----
            const char* gsrc = (const char*)(pk + (size_t)(ky * KH) * 4096);
#if USE_ASYNC_LDS
            {
                gv4i* gp = (gv4i*)gsrc;
                lv4i* lp = (lv4i*)smem;
                for (int i = threadIdx.x; i < KH * 512; i += 256) {
                    __builtin_amdgcn_global_load_async_to_lds_b128(
                        gp + i, lp + i, 0, 0);
                }
                __builtin_amdgcn_s_wait_asynccnt(0);
            }
#else
            for (int i = threadIdx.x; i < KH * 512; i += 256)
                ((uint4*)smem)[i] = ((const uint4*)gsrc)[i];
#endif
            __syncthreads();

            const int yy = y + ky - PAD;
            if (yy >= 0 && yy < H) {             // wave-uniform, no barrier inside
                for (int kx = 0; kx < KH; ++kx) {
                    // A fragment: row m = lane&15 -> pixel x0+m ; K = channel.
                    const int xx = x0 + n + kx - PAD;
                    v16bf a;
#pragma unroll
                    for (int e = 0; e < 16; ++e) a[e] = (__bf16)0.0f;
                    if (xx >= 0 && xx < W) {
                        const __bf16* q = img + ((size_t)yy * W + xx) * 32 + cb;
                        v8bf lo = *(const v8bf*)(q);        // K = cb .. cb+7
                        v8bf hh = *(const v8bf*)(q + 16);   // K = cb+16 .. cb+23
#pragma unroll
                        for (int e = 0; e < 8; ++e) { a[e] = lo[e]; a[e + 8] = hh[e]; }
                    }
                    // B fragments from LDS, double-buffered over the 8 N-tiles
                    const char* bp = smem + (size_t)kx * 8192 + (size_t)lane * 32;
                    v16bf bcur = *(const v16bf*)(bp);
#pragma unroll
                    for (int j = 0; j < 8; ++j) {
                        v16bf bnext = (j < 7) ? *(const v16bf*)(bp + (size_t)(j + 1) * 1024)
                                              : bcur;
                        acc[j] = __builtin_amdgcn_wmma_f32_16x16x32_bf16(
                            false, a, false, bcur, (short)0, acc[j], false, false);
                        bcur = bnext;
                    }
                }
            }
            __syncthreads();   // protect LDS before next ky-row overwrites it
        }
    }

    // LSTM cell update. Gate order (Keras): i | f | g | o across 128 channels.
    // Lane holds feature n (tiles 0,2,4,6) and n+16 (tiles 1,3,5,7),
    // rows m = 8*hi + r  ->  pixel pbase + m.
    const int prow = pbase + (hi << 3);
#pragma unroll
    for (int fs = 0; fs < 2; ++fs) {
        const int fidx = n + 16 * fs;
#pragma unroll
        for (int r = 0; r < 8; ++r) {
            const int p = prow + r;
            float zi = acc[0 + fs][r];
            float zf = acc[2 + fs][r];
            float zg = acc[4 + fs][r];
            float zo = acc[6 + fs][r];
            float ig = hard_sig(zi), fg = hard_sig(zf), og = hard_sig(zo);
            float gg = fmaxf(zg, 0.0f);
            float cold = cbuf[(size_t)fidx * M + p];
            float cn = fmaf(fg, cold, ig * gg);
            cbuf[(size_t)fidx * M + p] = cn;
            hout[(size_t)p * 32 + fidx] = (__bf16)(og * fmaxf(cn, 0.0f));
        }
    }
}

// ---------------------------------------------------------------------------
// BatchNorm (inference, eps=1e-3) + 2x2 maxpool. Writes step t of the next
// layer's input sequence (bf16 NHWC).
// ---------------------------------------------------------------------------
__global__ __launch_bounds__(256) void bnpool_kern(
    const __bf16* __restrict__ h,    // (B, 2Hn, 2Wn, 32)
    const float* __restrict__ g, const float* __restrict__ be,
    const float* __restrict__ mu, const float* __restrict__ var,
    __bf16* __restrict__ out,        // (B, 16, Hn, Wn, 32)
    int t, int Hn, int Wn)
{
    int idx = blockIdx.x * 256 + threadIdx.x;
    int total = 16 * Hn * Wn * 32;
    if (idx >= total) return;
    int f    = idx & 31;
    int rest = idx >> 5;
    int xo = rest % Wn; rest /= Wn;
    int yo = rest % Hn;
    int b  = rest / Hn;
    float sc = g[f] * rsqrtf(var[f] + 1e-3f);
    float sh = be[f] - mu[f] * sc;
    int W2 = Wn * 2;
    const __bf16* base = h + (((size_t)b * (Hn * 2) + yo * 2) * W2 + xo * 2) * 32 + f;
    float v00 = (float)base[0] * sc + sh;
    float v01 = (float)base[32] * sc + sh;
    float v10 = (float)base[(size_t)W2 * 32] * sc + sh;
    float v11 = (float)base[(size_t)W2 * 32 + 32] * sc + sh;
    float mx = fmaxf(fmaxf(v00, v01), fmaxf(v10, v11));
    out[(((size_t)(b * 16 + t) * Hn + yo) * Wn + xo) * 32 + f] = (__bf16)mx;
}

// ---------------------------------------------------------------------------
// GAP accumulation for one layer-3 timestep. One block per (b,f); fixed-order
// LDS tree reduction -> deterministic (no float atomics).
// ---------------------------------------------------------------------------
__global__ __launch_bounds__(256) void gap_accum_kern(
    const __bf16* __restrict__ h,    // (16, 16, 16, 32) = (B, 256 px, 32)
    float* __restrict__ gap)         // (16, 32), pre-zeroed
{
    __shared__ float s[256];
    int b = blockIdx.x >> 5;
    int f = blockIdx.x & 31;
    int i = threadIdx.x;
    s[i] = (float)h[(((size_t)b * 256) + i) * 32 + f];
    __syncthreads();
    for (int st = 128; st > 0; st >>= 1) {
        if (i < st) s[i] += s[i + st];
        __syncthreads();
    }
    if (i == 0) gap[b * 32 + f] += s[0];
}

// ---------------------------------------------------------------------------
// logits[b][n] = mean(gap)*wd + bd   (mean divisor T*H*W = 4096)
// ---------------------------------------------------------------------------
__global__ __launch_bounds__(256) void dense_kern(
    const float* __restrict__ gap, const float* __restrict__ wd,
    const float* __restrict__ bd, float* __restrict__ out)
{
    int idx = blockIdx.x * 256 + threadIdx.x;   // 16*4096
    int b = idx >> 12;
    int nn = idx & 4095;
    float s = bd[nn];
    const float scale = 1.0f / 4096.0f;
#pragma unroll
    for (int f = 0; f < 32; ++f)
        s = fmaf(gap[b * 32 + f] * scale, wd[(size_t)f * 4096 + nn], s);
    out[idx] = s;
}

__global__ __launch_bounds__(256) void softmax_kern(float* __restrict__ out)
{
    __shared__ float red[256];
    float* row = out + (size_t)blockIdx.x * 4096;
    int tid = threadIdx.x;
    float mx = -3.402823466e+38f;
    for (int i = tid; i < 4096; i += 256) mx = fmaxf(mx, row[i]);
    red[tid] = mx;
    __syncthreads();
    for (int st = 128; st > 0; st >>= 1) {
        if (tid < st) red[tid] = fmaxf(red[tid], red[tid + st]);
        __syncthreads();
    }
    mx = red[0];
    __syncthreads();
    float sum = 0.0f;
    for (int i = tid; i < 4096; i += 256) {
        float e = __expf(row[i] - mx);
        row[i] = e;
        sum += e;
    }
    red[tid] = sum;
    __syncthreads();
    for (int st = 128; st > 0; st >>= 1) {
        if (tid < st) red[tid] += red[tid + st];
        __syncthreads();
    }
    float inv = 1.0f / red[0];
    for (int i = tid; i < 4096; i += 256) row[i] *= inv;
}

// ---------------------------------------------------------------------------
extern "C" void kernel_launch(void* const* d_in, const int* in_sizes, int n_in,
                              void* d_out, int out_size, void* d_ws, size_t ws_size,
                              hipStream_t stream)
{
    (void)in_sizes; (void)n_in; (void)out_size; (void)ws_size;

    const float* x   = (const float*)d_in[0];
    const float* k1  = (const float*)d_in[1];
    const float* u1  = (const float*)d_in[2];
    const float* b1  = (const float*)d_in[3];
    const float* g1  = (const float*)d_in[4];
    const float* be1 = (const float*)d_in[5];
    const float* m1  = (const float*)d_in[6];
    const float* v1  = (const float*)d_in[7];
    const float* k2  = (const float*)d_in[8];
    const float* u2  = (const float*)d_in[9];
    const float* b2  = (const float*)d_in[10];
    const float* g2  = (const float*)d_in[11];
    const float* be2 = (const float*)d_in[12];
    const float* m2  = (const float*)d_in[13];
    const float* v2  = (const float*)d_in[14];
    const float* k3  = (const float*)d_in[15];
    const float* u3  = (const float*)d_in[16];
    const float* b3  = (const float*)d_in[17];
    const float* wd  = (const float*)d_in[18];
    const float* bd  = (const float*)d_in[19];
    float* out = (float*)d_out;

    // ---- workspace carve-up (all 256B aligned) ----
    char* base = (char*)d_ws;
    size_t off = 0;
    auto alloc = [&](size_t bytes) -> char* {
        char* p = base + off;
        off = (off + bytes + 255) & ~(size_t)255;
        return p;
    };
    const size_t M1 = 65536, M2 = 16384, M3 = 4096;
    __bf16* pU1 = (__bf16*)alloc((size_t)25 * 4096 * 2);
    __bf16* pW2 = (__bf16*)alloc((size_t)25 * 4096 * 2);
    __bf16* pU2 = (__bf16*)alloc((size_t)25 * 4096 * 2);
    __bf16* pW3 = (__bf16*)alloc((size_t)9 * 4096 * 2);
    __bf16* pU3 = (__bf16*)alloc((size_t)9 * 4096 * 2);
    __bf16* h1a = (__bf16*)alloc(M1 * 32 * 2);
    __bf16* h1b = (__bf16*)alloc(M1 * 32 * 2);
    float*  c1  = (float*)alloc(M1 * 32 * 4);
    __bf16* h2a = (__bf16*)alloc(M2 * 32 * 2);
    __bf16* h2b = (__bf16*)alloc(M2 * 32 * 2);
    float*  c2  = (float*)alloc(M2 * 32 * 4);
    __bf16* h3a = (__bf16*)alloc(M3 * 32 * 2);
    __bf16* h3b = (__bf16*)alloc(M3 * 32 * 2);
    float*  c3  = (float*)alloc(M3 * 32 * 4);
    __bf16* pooled1 = (__bf16*)alloc((size_t)16 * 16 * 32 * 32 * 32 * 2);
    __bf16* pooled2 = (__bf16*)alloc((size_t)16 * 16 * 16 * 16 * 32 * 2);
    float*  gap = (float*)alloc(16 * 32 * 4);

    // ---- zero recurrent state + GAP accumulator (deterministic per call) ----
    (void)hipMemsetAsync(h1a, 0, M1 * 32 * 2, stream);
    (void)hipMemsetAsync(c1,  0, M1 * 32 * 4, stream);
    (void)hipMemsetAsync(h2a, 0, M2 * 32 * 2, stream);
    (void)hipMemsetAsync(c2,  0, M2 * 32 * 4, stream);
    (void)hipMemsetAsync(h3a, 0, M3 * 32 * 2, stream);
    (void)hipMemsetAsync(c3,  0, M3 * 32 * 4, stream);
    (void)hipMemsetAsync(gap, 0, 16 * 32 * 4, stream);

    // ---- pack conv weights into WMMA B-fragment order (bf16) ----
    pack_b_kern<<<(25 * 4096 + 255) / 256, 256, 0, stream>>>(u1, pU1, 25);
    pack_b_kern<<<(25 * 4096 + 255) / 256, 256, 0, stream>>>(k2, pW2, 25);
    pack_b_kern<<<(25 * 4096 + 255) / 256, 256, 0, stream>>>(u2, pU2, 25);
    pack_b_kern<<<(9 * 4096 + 255) / 256, 256, 0, stream>>>(k3, pW3, 9);
    pack_b_kern<<<(9 * 4096 + 255) / 256, 256, 0, stream>>>(u3, pU3, 9);

    // ---- layer 1: 64x64, 5x5, Cin=1 ----
    {
        __bf16* hin = h1a; __bf16* hout = h1b;
        for (int t = 0; t < 16; ++t) {
            lstm_step_kern<64, 64, 5, true><<<512, 256, 0, stream>>>(
                nullptr, x, k1, nullptr, pU1, b1, hin, hout, c1, t);
            bnpool_kern<<<(16 * 32 * 32 * 32 + 255) / 256, 256, 0, stream>>>(
                hout, g1, be1, m1, v1, pooled1, t, 32, 32);
            __bf16* tmp = hin; hin = hout; hout = tmp;
        }
    }

    // ---- layer 2: 32x32, 5x5, Cin=32 ----
    {
        __bf16* hin = h2a; __bf16* hout = h2b;
        for (int t = 0; t < 16; ++t) {
            lstm_step_kern<32, 32, 5, false><<<128, 256, 0, stream>>>(
                pooled1, nullptr, nullptr, pW2, pU2, b2, hin, hout, c2, t);
            bnpool_kern<<<(16 * 16 * 16 * 32 + 255) / 256, 256, 0, stream>>>(
                hout, g2, be2, m2, v2, pooled2, t, 16, 16);
            __bf16* tmp = hin; hin = hout; hout = tmp;
        }
    }

    // ---- layer 3: 16x16, 3x3, Cin=32 + per-step GAP accumulation ----
    {
        __bf16* hin = h3a; __bf16* hout = h3b;
        for (int t = 0; t < 16; ++t) {
            lstm_step_kern<16, 16, 3, false><<<32, 256, 0, stream>>>(
                pooled2, nullptr, nullptr, pW3, pU3, b3, hin, hout, c3, t);
            gap_accum_kern<<<16 * 32, 256, 0, stream>>>(hout, gap);
            __bf16* tmp = hin; hin = hout; hout = tmp;
        }
    }

    // ---- dense + softmax ----
    dense_kern<<<256, 256, 0, stream>>>(gap, wd, bd, out);
    softmax_kern<<<16, 256, 0, stream>>>(out);
}
